// MemoryGate_61194694033999
// MI455X (gfx1250) — matrix operand order repaired and checked by reading, not compiled, not run on hardware
//
#include <hip/hip_runtime.h>

#define B_  4
#define N_  4096
#define M_  2048
#define D_  2048
#define E_  1024
#define K2_ 4096  // 2*D

typedef __attribute__((ext_vector_type(16))) __bf16          bf16x16;
typedef __attribute__((ext_vector_type(8)))  __bf16          bf16x8;
typedef __attribute__((ext_vector_type(16))) unsigned short  u16x16;
typedef __attribute__((ext_vector_type(8)))  unsigned short  u16x8;
typedef __attribute__((ext_vector_type(8)))  float           f32x8;
typedef int i32x4 __attribute__((vector_size(16)));

// ---- CDNA5 async global->LDS copy (ASYNCcnt path), with portable fallback ----
#if defined(__has_builtin)
#if __has_builtin(__builtin_amdgcn_global_load_async_to_lds_b128)
#define HAVE_ASYNC_LDS 1
#endif
#endif
#ifndef HAVE_ASYNC_LDS
#define HAVE_ASYNC_LDS 0
#endif

typedef __attribute__((address_space(1))) i32x4 as1_i32x4;
typedef __attribute__((address_space(3))) i32x4 as3_i32x4;

__device__ __forceinline__ void cp16_g2l(unsigned short* dst, const unsigned short* src) {
#if HAVE_ASYNC_LDS
  __builtin_amdgcn_global_load_async_to_lds_b128(
      (as1_i32x4*)(__attribute__((address_space(1))) void*)src,
      (as3_i32x4*)(__attribute__((address_space(3))) void*)dst, 0, 0);
#else
  *(u16x8*)dst = *(const u16x8*)src;
#endif
}
__device__ __forceinline__ void wait_async() {
#if HAVE_ASYNC_LDS
#if __has_builtin(__builtin_amdgcn_s_wait_asynccnt)
  __builtin_amdgcn_s_wait_asynccnt(0);
#else
  asm volatile("s_wait_asynccnt 0" ::: "memory");
#endif
#endif
}

// hardware f32->bf16 (packed cvt path)
__device__ __forceinline__ u16x8 pack8(const float* __restrict__ s) {
  f32x8 v = *(const f32x8*)s;
  return __builtin_bit_cast(u16x8, __builtin_convertvector(v, bf16x8));
}
__device__ __forceinline__ u16x8 pack8v(f32x8 v) {
  return __builtin_bit_cast(u16x8, __builtin_convertvector(v, bf16x8));
}

__device__ __forceinline__ f32x8 wmma_bf16(bf16x16 a, bf16x16 b, f32x8 c) {
  return __builtin_amdgcn_wmma_f32_16x16x32_bf16(false, a, false, b, (short)0, c,
                                                 false, false);
}

// ---------- fragment-order LDS addressing (contiguous k: lane half h holds k = 16h..16h+15) ----------
// Tiles stored as [tile][lane(32)][16 bf16]; fragment = one 32B load per lane.
__device__ __forceinline__ int a_slot(int tile, int rl, int kk) {
  int lane = rl + (((kk >> 4) & 1) << 4);
  int e0   = kk & 15;
  return ((tile * 32 + lane) << 4) + e0;
}
__device__ __forceinline__ int b_slot(int tile, int nl, int k0) {
  int lane = nl + (((k0 >> 4) & 1) << 4);
  int e0   = k0 & 15;
  return ((tile * 32 + lane) << 4) + e0;
}
__device__ __forceinline__ bf16x16 frag_lds(const unsigned short* base, int tile, int lane) {
  return __builtin_bit_cast(bf16x16, *(const u16x16*)(base + ((tile * 32 + lane) << 4)));
}
// B = Xᵀ where X row-major [n][k] bf16 in global: lane reads 16 contiguous bf16 (2x b128)
__device__ __forceinline__ bf16x16 load_b_frag_T(const unsigned short* x, int ld,
                                                 int nbase, int kbase, int lane) {
  const unsigned short* p =
      x + (long)(nbase + (lane & 15)) * ld + kbase + ((lane >> 4) << 4);
  return __builtin_bit_cast(bf16x16, *(const u16x16*)p);
}

// ============ Kernel 0: bulk f32 -> bf16 conversion (n8 groups of 8) ============
__global__ __launch_bounds__(256) void cvt_bf16_kernel(const float* __restrict__ in,
                                                       unsigned short* __restrict__ out,
                                                       long n8) {
  long i = (long)blockIdx.x * 256 + threadIdx.x;
  if (i < n8) *(u16x8*)(out + i * 8) = pack8(in + i * 8);
}

// ============ Kernel 1: memory_proj = memory @ W_memᵀ (writes proj and projT, bf16) ======
// 128x64 tile, K-slab 64 (2 k-chunks per stage -> 8 WMMAs between barriers)
__global__ __launch_bounds__(256) void proj_gemm_kernel(const unsigned short* __restrict__ memb,
                                                        const unsigned short* __restrict__ wmb,
                                                        unsigned short* __restrict__ proj,
                                                        unsigned short* __restrict__ projT) {
  __shared__ unsigned short sAf[16 * 32 * 16];  // [q(2)][wt(8)] frag-order tiles
  __shared__ unsigned short sBf[8 * 32 * 16];   // [q(2)][c(4)]
  const int b  = blockIdx.z;
  const int mb = blockIdx.y * 128;
  const int db = blockIdx.x * 64;
  const int t = threadIdx.x, lane = t & 31, w = t >> 5;
  const unsigned short* memB = memb + (long)b * M_ * E_;
  f32x8 acc[4];
#pragma unroll
  for (int c = 0; c < 4; ++c) acc[c] = (f32x8)0.f;

  for (int kb = 0; kb < E_; kb += 64) {
    {  // stage A 128x64 bf16 -> frag order (async g->LDS)
      int row = t >> 1, k0 = (t & 1) * 32;
      const unsigned short* src = memB + (long)(mb + row) * E_ + kb + k0;
      int wt = row >> 4, rl = row & 15;
#pragma unroll
      for (int ch = 0; ch < 4; ++ch) {
        int k = k0 + ch * 8;
        cp16_g2l(&sAf[a_slot(((k >> 5) << 3) + wt, rl, k & 31)], src + ch * 8);
      }
    }
    {  // stage B 64x64 from W_mem bf16 [d][e]
      int n = t >> 2, k0 = (t & 3) * 16;
      const unsigned short* src = wmb + (long)(db + n) * E_ + kb + k0;
      int c = n >> 4, nl = n & 15;
#pragma unroll
      for (int ch = 0; ch < 2; ++ch) {
        int k = k0 + ch * 8;
        cp16_g2l(&sBf[b_slot(((k >> 5) << 2) + c, nl, k & 31)], src + ch * 8);
      }
    }
    wait_async();
    __syncthreads();
#pragma unroll
    for (int q = 0; q < 2; ++q) {
      bf16x16 a = frag_lds(sAf, (q << 3) + w, lane);
#pragma unroll
      for (int c = 0; c < 4; ++c)
        acc[c] = wmma_bf16(a, frag_lds(sBf, (q << 2) + c, lane), acc[c]);
    }
    __syncthreads();
  }

  const int ncol = lane & 15, msel = (lane >> 4) * 8;
  const long mrow0 = mb + w * 16 + msel;
#pragma unroll
  for (int c = 0; c < 4; ++c) {
    long dcol = db + c * 16 + ncol;
    u16x8 pk = pack8v(acc[c]);
    *(u16x8*)&projT[(long)b * D_ * M_ + dcol * M_ + mrow0] = pk;
#pragma unroll
    for (int r = 0; r < 8; ++r)
      proj[(long)b * M_ * D_ + (mrow0 + r) * D_ + dcol] = pk[r];
  }
}

// ============ Kernel 2: fused attention for 16 query rows ============
__global__ __launch_bounds__(256) void attn_kernel(const unsigned short* __restrict__ hb,
                                                   const unsigned short* __restrict__ proj,
                                                   const unsigned short* __restrict__ projT,
                                                   float* __restrict__ ctx) {
  extern __shared__ char smemraw[];
  unsigned short* hqf = (unsigned short*)smemraw;                      // [64 tiles][32][16]
  float* sc     = (float*)(smemraw + 64 * 32 * 16 * 2);                // 16x2048 f32
  float* rowinv = (float*)(smemraw + 64 * 32 * 16 * 2 + 16 * 2048 * 4);

  const int b  = blockIdx.y;
  const int nb = blockIdx.x * 16;
  const int t = threadIdx.x, lane = t & 31, w = t >> 5;
  const unsigned short* projB  = proj  + (long)b * M_ * D_;
  const unsigned short* projTB = projT + (long)b * D_ * M_;

  {  // stage hidden 16x2048 bf16 -> frag order (async g->LDS)
    int row = t >> 4, c0 = (t & 15) * 128;
    const unsigned short* src = hb + ((long)b * N_ + nb + row) * D_ + c0;
#pragma unroll
    for (int g = 0; g < 16; ++g) {
      int k0 = c0 + g * 8;
      cp16_g2l(&hqf[a_slot(k0 >> 5, row, k0 & 31)], src + g * 8);
    }
  }
  wait_async();
  __syncthreads();

  {  // Phase A: scores = H @ projᵀ ; wave w owns m-columns [w*256, w*256+256)
    f32x8 acc[16];
#pragma unroll
    for (int mt = 0; mt < 16; ++mt) acc[mt] = (f32x8)0.f;
    for (int kt = 0; kt < 64; ++kt) {
      bf16x16 a = frag_lds(hqf, kt, lane);
#pragma unroll
      for (int mt = 0; mt < 16; ++mt) {
        bf16x16 bb = load_b_frag_T(projB, D_, w * 256 + mt * 16, kt * 32, lane);
        acc[mt] = wmma_bf16(a, bb, acc[mt]);
      }
    }
    const float scale = 0.022097086912079608f;  // 1/sqrt(2048)
    int ncol = lane & 15, msel = (lane >> 4) * 8;
#pragma unroll
    for (int mt = 0; mt < 16; ++mt)
#pragma unroll
      for (int r = 0; r < 8; ++r)
        sc[(msel + r) * 2048 + w * 256 + mt * 16 + ncol] = acc[mt][r] * scale;
  }
  __syncthreads();

  {  // softmax: 16 lanes per row, xor-shuffle reductions within half-wave
    int row = t >> 4, sub = t & 15;
    float* srow = sc + row * 2048;
    float mx = -3.4e38f;
    for (int i = sub; i < 2048; i += 16) mx = fmaxf(mx, srow[i]);
#pragma unroll
    for (int m = 1; m < 16; m <<= 1) mx = fmaxf(mx, __shfl_xor(mx, m, 32));
    float sum = 0.f;
    for (int i = sub; i < 2048; i += 16) {
      float e = __expf(srow[i] - mx);
      srow[i] = e;
      sum += e;
    }
#pragma unroll
    for (int m = 1; m < 16; m <<= 1) sum += __shfl_xor(sum, m, 32);
    if (sub == 0) rowinv[row] = 1.0f / sum;
  }
  __syncthreads();

  {  // convert P (1/rowsum folded) into bf16 frag order, reusing hqf
    int row = t >> 4, c0 = (t & 15) * 128;
    const float* srow = sc + row * 2048;
    float rinv = rowinv[row];
#pragma unroll
    for (int g = 0; g < 16; ++g) {
      int k0 = c0 + g * 8;
      f32x8 v = *(const f32x8*)(srow + k0) * rinv;
      *(u16x8*)&hqf[a_slot(k0 >> 5, row, k0 & 31)] = pack8v(v);
    }
  }
  __syncthreads();

  {  // Phase B: context = P @ proj ; wave w owns d-columns [w*256, w*256+256)
    f32x8 acc[16];
#pragma unroll
    for (int dt = 0; dt < 16; ++dt) acc[dt] = (f32x8)0.f;
    for (int kt = 0; kt < 64; ++kt) {
      bf16x16 a = frag_lds(hqf, kt, lane);
#pragma unroll
      for (int dt = 0; dt < 16; ++dt) {
        bf16x16 bb = load_b_frag_T(projTB, M_, w * 256 + dt * 16, kt * 32, lane);
        acc[dt] = wmma_bf16(a, bb, acc[dt]);
      }
    }
    int ncol = lane & 15, msel = (lane >> 4) * 8;
    float* cdst = ctx + ((long)b * N_ + nb) * D_;
#pragma unroll
    for (int dt = 0; dt < 16; ++dt)
#pragma unroll
      for (int r = 0; r < 8; ++r)
        cdst[(long)(msel + r) * D_ + w * 256 + dt * 16 + ncol] = acc[dt][r];
  }
}

// ============ Kernel 3: gate GEMM over K=4096 (concat, all-bf16 async staging) ============
__global__ __launch_bounds__(256) void gate_gemm_kernel(const unsigned short* __restrict__ hb,
                                                        const unsigned short* __restrict__ cb,
                                                        const float* __restrict__ hid,
                                                        const float* __restrict__ ctx,
                                                        const unsigned short* __restrict__ wgb,
                                                        const float* __restrict__ bg,
                                                        float* __restrict__ fused) {
  __shared__ unsigned short sAf[16 * 32 * 16];
  __shared__ unsigned short sBf[8 * 32 * 16];
  const int rb = blockIdx.y * 128;  // row in flattened [B*N]
  const int db = blockIdx.x * 64;
  const int t = threadIdx.x, lane = t & 31, w = t >> 5;
  f32x8 acc[4];
#pragma unroll
  for (int c = 0; c < 4; ++c) acc[c] = (f32x8)0.f;

  for (int kb = 0; kb < K2_; kb += 64) {
    {  // stage A: concat(hidden bf16, context bf16), fully async
      int row = t >> 1, k0 = (t & 1) * 32;
      long grow = rb + row;
      const unsigned short* src = (kb < D_) ? (hb + grow * D_ + kb + k0)
                                            : (cb + grow * D_ + (kb - D_) + k0);
      int wt = row >> 4, rl = row & 15;
#pragma unroll
      for (int ch = 0; ch < 4; ++ch) {
        int k = k0 + ch * 8;
        cp16_g2l(&sAf[a_slot(((k >> 5) << 3) + wt, rl, k & 31)], src + ch * 8);
      }
    }
    {  // stage B from bf16 W_gate[d][k]
      int n = t >> 2, k0 = (t & 3) * 16;
      const unsigned short* src = wgb + (long)(db + n) * K2_ + kb + k0;
      int c = n >> 4, nl = n & 15;
#pragma unroll
      for (int ch = 0; ch < 2; ++ch) {
        int k = k0 + ch * 8;
        cp16_g2l(&sBf[b_slot(((k >> 5) << 2) + c, nl, k & 31)], src + ch * 8);
      }
    }
    wait_async();
    __syncthreads();
#pragma unroll
    for (int q = 0; q < 2; ++q) {
      bf16x16 a = frag_lds(sAf, (q << 3) + w, lane);
#pragma unroll
      for (int c = 0; c < 4; ++c)
        acc[c] = wmma_bf16(a, frag_lds(sBf, (q << 2) + c, lane), acc[c]);
    }
    __syncthreads();
  }

  int ncol = lane & 15, msel = (lane >> 4) * 8;
#pragma unroll
  for (int c = 0; c < 4; ++c) {
    int dcol = db + c * 16 + ncol;
    float bias = bg[dcol];
#pragma unroll
    for (int r = 0; r < 8; ++r) {
      long grow = rb + w * 16 + msel + r;
      float g  = 1.0f / (1.0f + __expf(-(acc[c][r] + bias)));
      float h  = hid[grow * D_ + dcol];
      float cv = ctx[grow * D_ + dcol];
      fused[grow * D_ + dcol] = h + g * cv;
    }
  }
}

// ============ Kernel 4: RMSNorm, one block per row ============
__global__ __launch_bounds__(256) void rmsnorm_kernel(const float* __restrict__ fused,
                                                      const float* __restrict__ nw,
                                                      float* __restrict__ out) {
  __shared__ float red[8];
  long row = blockIdx.x;
  const float* src = fused + row * D_;
  float* dst = out + row * D_;
  int t = threadIdx.x;
  float v[8];
  float ss = 0.f;
#pragma unroll
  for (int i = 0; i < 8; ++i) {
    v[i] = src[t + i * 256];
    ss += v[i] * v[i];
  }
#pragma unroll
  for (int m = 1; m < 32; m <<= 1) ss += __shfl_xor(ss, m, 32);
  if ((t & 31) == 0) red[t >> 5] = ss;
  __syncthreads();
  float tot = 0.f;
#pragma unroll
  for (int i = 0; i < 8; ++i) tot += red[i];
  float rs = rsqrtf(tot / (float)D_ + 1e-6f);
#pragma unroll
  for (int i = 0; i < 8; ++i) dst[t + i * 256] = v[i] * rs * nw[t + i * 256];
}

extern "C" void kernel_launch(void* const* d_in, const int* in_sizes, int n_in,
                              void* d_out, int out_size, void* d_ws, size_t ws_size,
                              hipStream_t stream) {
  (void)in_sizes; (void)n_in; (void)out_size; (void)ws_size;
  const float* hid = (const float*)d_in[0];
  const float* mem = (const float*)d_in[1];
  const float* Wm  = (const float*)d_in[2];
  const float* Wg  = (const float*)d_in[3];
  const float* bg  = (const float*)d_in[4];
  const float* nw  = (const float*)d_in[5];
  float* out = (float*)d_out;

  char* ws = (char*)d_ws;
  unsigned short* proj  = (unsigned short*)ws;                 //  33,554,432 B  bf16 [b][m][d]
  unsigned short* projT = (unsigned short*)(ws + 33554432);    //  33,554,432 B  bf16 [b][d][m]
  float* ctx   = (float*)(ws + 67108864);                      // 134,217,728 B
  float* fused = (float*)(ws + 201326592);                     // 134,217,728 B
  unsigned short* hb16  = (unsigned short*)(ws + 335544320);   //  67,108,864 B  bf16 hidden
  unsigned short* memb16= (unsigned short*)(ws + 402653184);   //  16,777,216 B  bf16 memory
  unsigned short* wmb16 = (unsigned short*)(ws + 419430400);   //   4,194,304 B  bf16 W_mem
  unsigned short* wgb16 = (unsigned short*)(ws + 423624704);   //  16,777,216 B  bf16 W_gate
  unsigned short* ctxb16= (unsigned short*)(ws + 440401920);   //  67,108,864 B  bf16 context

  // bulk conversions (grid = elems/8/256)
  cvt_bf16_kernel<<<16384, 256, 0, stream>>>(hid, hb16, 4194304);
  cvt_bf16_kernel<<<4096,  256, 0, stream>>>(mem, memb16, 1048576);
  cvt_bf16_kernel<<<1024,  256, 0, stream>>>(Wm,  wmb16,  262144);
  cvt_bf16_kernel<<<4096,  256, 0, stream>>>(Wg,  wgb16, 1048576);

  proj_gemm_kernel<<<dim3(D_ / 64, M_ / 128, B_), 256, 0, stream>>>(memb16, wmb16, proj, projT);
  attn_kernel<<<dim3(N_ / 16, B_), 256, 196672, stream>>>(hb16, proj, projT, ctx);
  cvt_bf16_kernel<<<16384, 256, 0, stream>>>(ctx, ctxb16, 4194304);
  gate_gemm_kernel<<<dim3(D_ / 64, (B_ * N_) / 128), 256, 0, stream>>>(hb16, ctxb16, hid, ctx,
                                                                       wgb16, bg, fused);
  rmsnorm_kernel<<<B_ * N_, 256, 0, stream>>>(fused, nw, out);
}
